// SRNO_20486994002448
// MI455X (gfx1250) — matrix operand order repaired and checked by reading, not compiled, not run on hardware
//
#include <hip/hip_runtime.h>
#include <math.h>

typedef __bf16 bf16_t;
typedef __attribute__((ext_vector_type(16))) __bf16 v16bf;
typedef __attribute__((ext_vector_type(8)))  float  v8f;

#define DIVUP(a,b) (((a)+(b)-1)/(b))

// gfx1250 async global->LDS staging (ASYNCcnt path), if the toolchain has it
#if defined(__has_builtin)
#if __has_builtin(__builtin_amdgcn_global_load_async_to_lds_b128) && \
    __has_builtin(__builtin_amdgcn_s_wait_asynccnt)
#define USE_ASYNC_LDS 1
#endif
#endif

#ifdef USE_ASYNC_LDS
typedef int vi4_t __attribute__((vector_size(16)));
__device__ __forceinline__ void async_cp16(const bf16_t* g, bf16_t* l) {
  __builtin_amdgcn_global_load_async_to_lds_b128(
      (__attribute__((address_space(1))) vi4_t*)(g),
      (__attribute__((address_space(3))) vi4_t*)(l),
      0, 0);
}
#endif

__device__ __forceinline__ bf16_t f2b(float f) {
  unsigned u = __float_as_uint(f);
  u = (u + 0x7FFFu + ((u >> 16) & 1u)) >> 16;
  unsigned short s = (unsigned short)u;
  return __builtin_bit_cast(bf16_t, s);
}
__device__ __forceinline__ float b2f(bf16_t b) {
  unsigned short s = __builtin_bit_cast(unsigned short, b);
  return __uint_as_float(((unsigned)s) << 16);
}
__device__ __forceinline__ float gelu_f(float x) {
  return 0.5f * x * (1.0f + erff(x * 0.70710678118654752f));
}

// ---------------------------------------------------------------------------
// Fragment helpers (documented CDNA5 wave32 WMMA register layouts)
//   A 16x32 bf16: lane = kHalf*16 + M ; two contiguous 16B runs per lane
//   B 32x16 bf16: lane = kHalf*16 + N ; same K order as A (strided by ldb)
//   C 16x16 f32 : VGPR r -> M = r + 8*(lane>=16), N = lane&15
// ---------------------------------------------------------------------------
__device__ __forceinline__ v16bf frag_a(const bf16_t* row /* &As[m*lda + ksub] */) {
  const int kh = ((threadIdx.x & 31) >> 4) * 8;
  v16bf a;
#pragma unroll
  for (int e = 0; e < 8; ++e) { a[e] = row[kh + e]; a[e + 8] = row[16 + kh + e]; }
  return a;
}
__device__ __forceinline__ v16bf frag_b(const bf16_t* p /* &Bs[ksub*ldb + n] */, int ldb) {
  const int kh = ((threadIdx.x & 31) >> 4) * 8;
  v16bf b;
#pragma unroll
  for (int e = 0; e < 8; ++e) { b[e] = p[(kh + e) * ldb]; b[e + 8] = p[(16 + kh + e) * ldb]; }
  return b;
}

// ---------------------------------------------------------------------------
// Dense WMMA GEMM: C[M x Ntot] = A[M x K] * B[K x Ntot]
//   BLOCK 128M x 128N x 64K, 8 waves of 32x64 (2x4 wmma tiles, 16 wmma/iter)
//   epilogue: +bias; act 0=none 1=gelu 2=relu; +res(f32); writes f32/bf16
// ---------------------------------------------------------------------------
#define LDA_D 72    // 64 + 8 pad (bf16 elems)
#define LDB_D 136   // 128 + 8 pad

__global__ __launch_bounds__(256)
void gemm_dense(const bf16_t* __restrict__ A, const bf16_t* __restrict__ Bm,
                const float* __restrict__ bias, const float* __restrict__ res,
                float* __restrict__ outf, bf16_t* __restrict__ outb,
                int M, int Ntot, int K, int act)
{
  __shared__ bf16_t As[128 * LDA_D];
  __shared__ bf16_t Bs[64 * LDB_D];
  const int tid  = threadIdx.x;
  const int lane = tid & 31;
  const int wave = tid >> 5;
  const int wm = wave & 3;        // 4 waves along M (32 each)
  const int wn = wave >> 2;       // 2 waves along N (64 each)
  const int m0 = blockIdx.x * 128;
  const int n0 = blockIdx.y * 128;

  v8f acc[2][4];
#pragma unroll
  for (int i = 0; i < 2; ++i)
#pragma unroll
    for (int j = 0; j < 4; ++j)
#pragma unroll
      for (int e = 0; e < 8; ++e) acc[i][j][e] = 0.0f;

  const int ar = tid >> 1;            // A stager: row 0..127, 32 bf16 each
  const int ac = (tid & 1) * 32;
  const int br = tid >> 2;            // B stager: row 0..63, 32 bf16 each
  const int bs = (tid & 3) * 32;

  for (int k0 = 0; k0 < K; k0 += 64) {
    const bf16_t* ga = A  + ((size_t)(m0 + ar) * K + k0 + ac);
    const bf16_t* gb = Bm + ((size_t)(k0 + br) * Ntot + n0 + bs);
    bf16_t* la = &As[ar * LDA_D + ac];
    bf16_t* lb = &Bs[br * LDB_D + bs];
#ifdef USE_ASYNC_LDS
#pragma unroll
    for (int q = 0; q < 4; ++q) {
      async_cp16(ga + q * 8, la + q * 8);
      async_cp16(gb + q * 8, lb + q * 8);
    }
    __builtin_amdgcn_s_wait_asynccnt(0);
#else
#pragma unroll
    for (int q = 0; q < 4; ++q) {
      *(uint4*)(la + q * 8) = *(const uint4*)(ga + q * 8);
      *(uint4*)(lb + q * 8) = *(const uint4*)(gb + q * 8);
    }
    if (k0 + 64 < K) __builtin_prefetch(ga + 64, 0, 1);
#endif
    __syncthreads();

#pragma unroll
    for (int ks = 0; ks < 2; ++ks) {
      v16bf af[2], bf[4];
#pragma unroll
      for (int mi = 0; mi < 2; ++mi)
        af[mi] = frag_a(&As[(wm * 32 + mi * 16 + (lane & 15)) * LDA_D + ks * 32]);
#pragma unroll
      for (int ni = 0; ni < 4; ++ni)
        bf[ni] = frag_b(&Bs[(ks * 32) * LDB_D + wn * 64 + ni * 16 + (lane & 15)], LDB_D);
#pragma unroll
      for (int mi = 0; mi < 2; ++mi)
#pragma unroll
        for (int ni = 0; ni < 4; ++ni)
          acc[mi][ni] = __builtin_amdgcn_wmma_f32_16x16x32_bf16(
              false, af[mi], false, bf[ni], (short)0, acc[mi][ni], false, false);
    }
    __syncthreads();
  }

#pragma unroll
  for (int mi = 0; mi < 2; ++mi) {
#pragma unroll
    for (int ni = 0; ni < 4; ++ni) {
      const int nc = n0 + wn * 64 + ni * 16 + (lane & 15);
      const float bval = bias ? bias[nc] : 0.0f;
#pragma unroll
      for (int r = 0; r < 8; ++r) {
        const int mm = m0 + wm * 32 + mi * 16 + r + 8 * (lane >> 4);
        float v = acc[mi][ni][r] + bval;
        if (act == 1) v = gelu_f(v);
        else if (act == 2) v = fmaxf(v, 0.0f);
        const size_t off = (size_t)mm * Ntot + nc;
        if (res)  v += res[off];
        if (outf) outf[off] = v;
        if (outb) outb[off] = f2b(v);
      }
    }
  }
}

// ---------------------------------------------------------------------------
// Conv3x3 implicit-GEMM WMMA (64->64 ch, 64x64 NHWC bf16 feature map)
//   BLOCK 128M x 64N x 32K, 8 waves of 32x32; K = tap*64 + channel (576)
// ---------------------------------------------------------------------------
#define LDA_C 40   // 32 + 8 pad
#define LDB_C 72   // 64 + 8 pad

__global__ __launch_bounds__(256)
void conv3_wmma(const bf16_t* __restrict__ A, const bf16_t* __restrict__ Bm,
                const float* __restrict__ bias, const float* __restrict__ res,
                float* __restrict__ outf, bf16_t* __restrict__ outb, int act)
{
  __shared__ bf16_t As[128 * LDA_C];
  __shared__ bf16_t Bs[32 * LDB_C];
  const int tid  = threadIdx.x;
  const int lane = tid & 31;
  const int wave = tid >> 5;
  const int wm = wave & 3;
  const int wn = wave >> 2;
  const int m0 = blockIdx.x * 128;

  v8f acc[2][2];
#pragma unroll
  for (int i = 0; i < 2; ++i)
#pragma unroll
    for (int j = 0; j < 2; ++j)
#pragma unroll
      for (int e = 0; e < 8; ++e) acc[i][j][e] = 0.0f;

  const int ar = tid >> 1;
  const int ac = (tid & 1) * 16;
  const int br = tid >> 3;
  const int bc = (tid & 7) * 8;

  for (int k0 = 0; k0 < 576; k0 += 32) {
    uint4 av0, av1;
    {
      int tap = k0 >> 6;
      int c0  = (k0 & 63) + ac;
      int p = m0 + ar;
      int y = (p >> 6) + (tap / 3) - 1;
      int x = (p & 63) + (tap % 3) - 1;
      if (y >= 0 && y < 64 && x >= 0 && x < 64) {
        const uint4* src = (const uint4*)(A + ((size_t)(y * 64 + x) * 64 + c0));
        av0 = src[0]; av1 = src[1];
      } else {
        av0 = uint4{0,0,0,0}; av1 = uint4{0,0,0,0};
      }
    }
    *(uint4*)(&As[ar * LDA_C + ac])     = av0;
    *(uint4*)(&As[ar * LDA_C + ac + 8]) = av1;
    *(uint4*)(&Bs[br * LDB_C + bc]) =
        *(const uint4*)(Bm + ((size_t)(k0 + br) * 64 + bc));
    __syncthreads();

    v16bf af[2], bf[2];
#pragma unroll
    for (int mi = 0; mi < 2; ++mi)
      af[mi] = frag_a(&As[(wm * 32 + mi * 16 + (lane & 15)) * LDA_C]);
#pragma unroll
    for (int ni = 0; ni < 2; ++ni)
      bf[ni] = frag_b(&Bs[wn * 32 + ni * 16 + (lane & 15)], LDB_C);
#pragma unroll
    for (int mi = 0; mi < 2; ++mi)
#pragma unroll
      for (int ni = 0; ni < 2; ++ni)
        acc[mi][ni] = __builtin_amdgcn_wmma_f32_16x16x32_bf16(
            false, af[mi], false, bf[ni], (short)0, acc[mi][ni], false, false);
    __syncthreads();
  }

#pragma unroll
  for (int mi = 0; mi < 2; ++mi) {
#pragma unroll
    for (int ni = 0; ni < 2; ++ni) {
      const int nc = wn * 32 + ni * 16 + (lane & 15);
      const float bval = bias ? bias[nc] : 0.0f;
#pragma unroll
      for (int r = 0; r < 8; ++r) {
        const int mm = m0 + wm * 32 + mi * 16 + r + 8 * (lane >> 4);
        float v = acc[mi][ni][r] + bval;
        if (act == 2) v = fmaxf(v, 0.0f);
        const size_t off = (size_t)mm * 64 + nc;
        if (res)  v += res[off];
        if (outf) outf[off] = v;
        if (outb) outb[off] = f2b(v);
      }
    }
  }
}

// ---------------------------------------------------------------------------
// Weight prep: f32 [N,K] -> bf16 B[kPad][N] (W^T, zero padded K)
// ---------------------------------------------------------------------------
__global__ void prep_dense(const float* __restrict__ W, bf16_t* __restrict__ Bm,
                           int K, int Kpad, int N)
{
  int idx = blockIdx.x * 256 + threadIdx.x;
  if (idx >= Kpad * N) return;
  int k = idx / N, n = idx % N;
  Bm[idx] = (k < K) ? f2b(W[(size_t)n * K + k]) : f2b(0.0f);
}
// conv weight f32 [64][64][3][3] -> bf16 B[(tap*64+c)][o] (576 x 64)
__global__ void prep_conv(const float* __restrict__ W, bf16_t* __restrict__ Bm)
{
  int idx = blockIdx.x * 256 + threadIdx.x;
  if (idx >= 576 * 64) return;
  int k = idx >> 6, o = idx & 63;
  int tap = k >> 6, c = k & 63;
  Bm[idx] = f2b(W[(size_t)o * 576 + c * 9 + tap]);
}

// ---------------------------------------------------------------------------
// EDSR head conv3x3: 1 -> 64 channels, writes h(f32), y(f32), y(bf16) NHWC
// ---------------------------------------------------------------------------
__global__ void head_conv(const float* __restrict__ inp, const float* __restrict__ w,
                          const float* __restrict__ b,
                          float* __restrict__ hf, float* __restrict__ yf, bf16_t* __restrict__ yb)
{
  int idx = blockIdx.x * 256 + threadIdx.x;
  if (idx >= 4096 * 64) return;
  int p = idx >> 6, ch = idx & 63;
  int y = p >> 6, x = p & 63;
  float acc = b[ch];
#pragma unroll
  for (int dy = 0; dy < 3; ++dy)
#pragma unroll
    for (int dx = 0; dx < 3; ++dx) {
      int ys = y + dy - 1, xs = x + dx - 1;
      if (ys >= 0 && ys < 64 && xs >= 0 && xs < 64)
        acc += w[ch * 9 + dy * 3 + dx] * inp[ys * 64 + xs];
    }
  hf[idx] = acc; yf[idx] = acc; yb[idx] = f2b(acc);
}

// ---------------------------------------------------------------------------
// Grid assembly: 4 nearest samples + area weights + rel coords + cell
// grid is [65536][320] bf16 (266 real channels, zero pad to 320)
// ---------------------------------------------------------------------------
__global__ __launch_bounds__(256)
void grid_assemble(const float* __restrict__ feat, bf16_t* __restrict__ grid)
{
  const int wave = threadIdx.x >> 5, lane = threadIdx.x & 31;
  const int p = blockIdx.x * 8 + wave;          // 0..65535
  const int i = p >> 8, j = p & 255;
  const float c0b = -1.0f + (2.0f * (float)i + 1.0f) / 256.0f;
  const float c1b = -1.0f + (2.0f * (float)j + 1.0f) / 256.0f;
  const float r = 1.0f / 64.0f;
  int ihv[4], iwv[4];
  float r0v[4], r1v[4], area[4];
#pragma unroll
  for (int s = 0; s < 4; ++s) {
    float vx = (s < 2) ? -1.0f : 1.0f;
    float vy = (s & 1) ? 1.0f : -1.0f;
    float c0 = fminf(fmaxf(c0b + vx * r + 1e-6f, -1.0f + 1e-6f), 1.0f - 1e-6f);
    float c1 = fminf(fmaxf(c1b + vy * r + 1e-6f, -1.0f + 1e-6f), 1.0f - 1e-6f);
    int ih = (int)fminf(fmaxf(rintf(((c0 + 1.0f) * 64.0f - 1.0f) * 0.5f), 0.0f), 63.0f);
    int iw = (int)fminf(fmaxf(rintf(((c1 + 1.0f) * 64.0f - 1.0f) * 0.5f), 0.0f), 63.0f);
    float o0 = -1.0f + (2.0f * (float)ih + 1.0f) / 64.0f;
    float o1 = -1.0f + (2.0f * (float)iw + 1.0f) / 64.0f;
    float r0 = (c0b - o0) * 64.0f;
    float r1 = (c1b - o1) * 64.0f;
    ihv[s] = ih; iwv[s] = iw; r0v[s] = r0; r1v[s] = r1;
    area[s] = fabsf(r0 * r1) + 1e-9f;
  }
  const float tot = area[0] + area[1] + area[2] + area[3];
  const size_t base = (size_t)p * 320;
#pragma unroll
  for (int s = 0; s < 4; ++s) {
    float wgt = area[3 - s] / tot;    // diagonal-swapped areas
    const float* fp = feat + (size_t)(ihv[s] * 64 + iwv[s]) * 64;
    grid[base + 8 + s * 64 + lane]      = f2b(fp[lane] * wgt);
    grid[base + 8 + s * 64 + lane + 32] = f2b(fp[lane + 32] * wgt);
  }
  if (lane < 8) {
    int s = lane >> 1;
    grid[base + lane] = f2b((lane & 1) ? r1v[s] : r0v[s]);
  }
  if (lane == 8)  grid[base + 264] = f2b(0.5f);   // cell: 2/Ho*h
  if (lane == 9)  grid[base + 265] = f2b(0.5f);   // cell: 2/Wo*w
  grid[base + 266 + lane] = f2b(0.0f);            // K pad 266..297
  if (lane < 22) grid[base + 298 + lane] = f2b(0.0f);   // K pad 298..319
}

// ---------------------------------------------------------------------------
// LayerNorm of k and v per (pixel, head), hc=16; qkv channel = h*48 + {q,k,v}
// ---------------------------------------------------------------------------
__global__ void ln_kv(const bf16_t* __restrict__ qkv,
                      const float* __restrict__ kg, const float* __restrict__ kb,
                      const float* __restrict__ vg, const float* __restrict__ vb,
                      bf16_t* __restrict__ kout, bf16_t* __restrict__ vout)
{
  int idx = blockIdx.x * 256 + threadIdx.x;     // 65536*16
  if (idx >= 65536 * 16) return;
  int n = idx >> 4, h = idx & 15;
  const bf16_t* kp = qkv + (size_t)n * 768 + h * 48 + 16;
  const bf16_t* vp = kp + 16;
  float kv_[16], vv_[16], km = 0.0f, vm = 0.0f;
#pragma unroll
  for (int c = 0; c < 16; ++c) { kv_[c] = b2f(kp[c]); vv_[c] = b2f(vp[c]); km += kv_[c]; vm += vv_[c]; }
  km *= (1.0f / 16.0f); vm *= (1.0f / 16.0f);
  float kvar = 0.0f, vvar = 0.0f;
#pragma unroll
  for (int c = 0; c < 16; ++c) {
    float a = kv_[c] - km; kvar += a * a;
    float b = vv_[c] - vm; vvar += b * b;
  }
  float kr = rsqrtf(kvar * (1.0f / 16.0f) + 1e-5f);
  float vr = rsqrtf(vvar * (1.0f / 16.0f) + 1e-5f);
  bf16_t* ko = kout + (size_t)n * 256 + h * 16;
  bf16_t* vo = vout + (size_t)n * 256 + h * 16;
#pragma unroll
  for (int c = 0; c < 16; ++c) {
    ko[c] = f2b((kv_[c] - km) * kr * kg[h * 16 + c] + kb[h * 16 + c]);
    vo[c] = f2b((vv_[c] - vm) * vr * vg[h * 16 + c] + vb[h * 16 + c]);
  }
}

// ---------------------------------------------------------------------------
// kv[h][c][d] = (1/N) sum_n k[n,h,c]*v[n,h,d] : partial over 1024-pixel chunks
// ---------------------------------------------------------------------------
__global__ __launch_bounds__(256)
void kv_partial(const bf16_t* __restrict__ kln, const bf16_t* __restrict__ vln,
                float* __restrict__ part)
{
  __shared__ bf16_t kl[64 * 16];
  __shared__ bf16_t vl[64 * 16];
  const int head = blockIdx.x >> 6;
  const int chunk = blockIdx.x & 63;
  const int tid = threadIdx.x;
  const int c = tid >> 4, d = tid & 15;
  const int pl = tid >> 2, c4 = (tid & 3) * 4;
  float acc = 0.0f;
  for (int s = 0; s < 16; ++s) {
    const int pb = chunk * 1024 + s * 64;
    *(uint2*)(&kl[pl * 16 + c4]) = *(const uint2*)(kln + (size_t)(pb + pl) * 256 + head * 16 + c4);
    *(uint2*)(&vl[pl * 16 + c4]) = *(const uint2*)(vln + (size_t)(pb + pl) * 256 + head * 16 + c4);
    __syncthreads();
#pragma unroll 8
    for (int n = 0; n < 64; ++n)
      acc += b2f(kl[n * 16 + c]) * b2f(vl[n * 16 + d]);
    __syncthreads();
  }
  part[(size_t)blockIdx.x * 256 + tid] = acc;
}

__global__ void kv_finalize(const float* __restrict__ part, float* __restrict__ kv)
{
  const int head = blockIdx.x;       // 16
  const int tid = threadIdx.x;       // 256 = c*16+d
  float s = 0.0f;
  for (int c = 0; c < 64; ++c) s += part[(size_t)(head * 64 + c) * 256 + tid];
  kv[head * 256 + tid] = s * (1.0f / 65536.0f);
}

// ---------------------------------------------------------------------------
// ret[n, h*16+d] = sum_c q[n,h,c]*kv[h,c,d] + x[n, h*16+d]
// ---------------------------------------------------------------------------
__global__ __launch_bounds__(256)
void attn_apply(const bf16_t* __restrict__ qkv, const float* __restrict__ kv,
                const float* __restrict__ xin,
                float* __restrict__ retf, bf16_t* __restrict__ retb)
{
  __shared__ float kvs[16 * 256];
  const int tid = threadIdx.x;
#pragma unroll
  for (int i = 0; i < 16; ++i) kvs[i * 256 + tid] = kv[i * 256 + tid];
  __syncthreads();
  const int pair = blockIdx.x * 256 + tid;   // (pixel, head)
  const int n = pair >> 4, h = pair & 15;
  float q[16];
  const bf16_t* qp = qkv + (size_t)n * 768 + h * 48;
#pragma unroll
  for (int c = 0; c < 16; ++c) q[c] = b2f(qp[c]);
  const float* kvh = kvs + h * 256;
#pragma unroll
  for (int d = 0; d < 16; ++d) {
    float acc = 0.0f;
#pragma unroll
    for (int c = 0; c < 16; ++c) acc += q[c] * kvh[c * 16 + d];
    const size_t off = (size_t)n * 256 + h * 16 + d;
    float v = acc + xin[off];
    retf[off] = v; retb[off] = f2b(v);
  }
}

// ---------------------------------------------------------------------------
// out[p] = dot(mid[p,:], fc2_w) + fc2_b + bilinear_border(inp, p)
// ---------------------------------------------------------------------------
__global__ __launch_bounds__(256)
void final_fc2(const bf16_t* __restrict__ mid, const float* __restrict__ w2,
               const float* __restrict__ b2, const float* __restrict__ inp,
               float* __restrict__ out)
{
  const int wave = threadIdx.x >> 5, lane = threadIdx.x & 31;
  const int p = blockIdx.x * 8 + wave;
  const bf16_t* mp = mid + (size_t)p * 256 + lane * 8;
  float partial = 0.0f;
#pragma unroll
  for (int i = 0; i < 8; ++i) partial += b2f(mp[i]) * w2[lane * 8 + i];
#pragma unroll
  for (int off = 16; off > 0; off >>= 1) partial += __shfl_down(partial, off, 32);
  if (lane == 0) {
    const int i = p >> 8, j = p & 255;
    float cy = -1.0f + (2.0f * (float)i + 1.0f) / 256.0f;
    float cx = -1.0f + (2.0f * (float)j + 1.0f) / 256.0f;
    float iy = fminf(fmaxf(((cy + 1.0f) * 64.0f - 1.0f) * 0.5f, 0.0f), 63.0f);
    float ix = fminf(fmaxf(((cx + 1.0f) * 64.0f - 1.0f) * 0.5f, 0.0f), 63.0f);
    float y0f = floorf(iy), x0f = floorf(ix);
    float wy = iy - y0f, wx = ix - x0f;
    int y0 = (int)y0f, x0 = (int)x0f;
    int y1 = (y0 + 1 < 63) ? y0 + 1 : 63;
    int x1 = (x0 + 1 < 63) ? x0 + 1 : 63;
    float bl = inp[y0 * 64 + x0] * (1.0f - wy) * (1.0f - wx)
             + inp[y0 * 64 + x1] * (1.0f - wy) * wx
             + inp[y1 * 64 + x0] * wy * (1.0f - wx)
             + inp[y1 * 64 + x1] * wy * wx;
    out[p] = partial + b2[0] + bl;
  }
}

// ---------------------------------------------------------------------------
extern "C" void kernel_launch(void* const* d_in, const int* in_sizes, int n_in,
                              void* d_out, int out_size, void* d_ws, size_t ws_size,
                              hipStream_t stream)
{
  (void)in_sizes; (void)n_in; (void)out_size; (void)ws_size;
  const float* inp    = (const float*)d_in[0];
  const float* head_w = (const float*)d_in[1];
  const float* head_b = (const float*)d_in[2];
  const float* res_w1 = (const float*)d_in[3];
  const float* res_b1 = (const float*)d_in[4];
  const float* res_w2 = (const float*)d_in[5];
  const float* res_b2 = (const float*)d_in[6];
  const float* tail_w = (const float*)d_in[7];
  const float* tail_b = (const float*)d_in[8];
  const float* c00_w  = (const float*)d_in[9];
  const float* c00_b  = (const float*)d_in[10];
  const float* fc1_w  = (const float*)d_in[11];
  const float* fc1_b  = (const float*)d_in[12];
  const float* fc2_w  = (const float*)d_in[13];
  const float* fc2_b  = (const float*)d_in[14];
  const float* attn[2][10];
  for (int a = 0; a < 2; ++a)
    for (int i = 0; i < 10; ++i)
      attn[a][i] = (const float*)d_in[15 + a * 10 + i];
  float* out = (float*)d_out;

  char* ws = (char*)d_ws;
  size_t off = 0;
  auto alloc = [&](size_t bytes) -> void* {
    void* p = ws + off;
    off += (bytes + 255) & ~(size_t)255;
    return p;
  };
  const int NP = 65536;   // 256x256 output pixels
  const int FP = 4096;    // 64x64 feature pixels

  float*  h_f    = (float*) alloc((size_t)FP * 64 * 4);
  float*  y_f    = (float*) alloc((size_t)FP * 64 * 4);
  bf16_t* y_b    = (bf16_t*)alloc((size_t)FP * 64 * 2);
  bf16_t* z_b    = (bf16_t*)alloc((size_t)FP * 64 * 2);
  float*  feat_f = (float*) alloc((size_t)FP * 64 * 4);
  bf16_t* Bw_r1  = (bf16_t*)alloc((size_t)16 * 576 * 64 * 2);
  bf16_t* Bw_r2  = (bf16_t*)alloc((size_t)16 * 576 * 64 * 2);
  bf16_t* Bw_tl  = (bf16_t*)alloc((size_t)576 * 64 * 2);
  bf16_t* grid_b = (bf16_t*)alloc((size_t)NP * 320 * 2);
  bf16_t* Bw_c00 = (bf16_t*)alloc((size_t)320 * 256 * 2);
  float*  x0_f   = (float*) alloc((size_t)NP * 256 * 4);
  bf16_t* x0_b   = (bf16_t*)alloc((size_t)NP * 256 * 2);
  float*  x1_f   = (float*) alloc((size_t)NP * 256 * 4);
  bf16_t* x1_b   = (bf16_t*)alloc((size_t)NP * 256 * 2);
  bf16_t* qkv_b  = (bf16_t*)alloc((size_t)NP * 768 * 2);
  bf16_t* kln_b  = (bf16_t*)alloc((size_t)NP * 256 * 2);
  bf16_t* vln_b  = (bf16_t*)alloc((size_t)NP * 256 * 2);
  float*  kv_p   = (float*) alloc((size_t)16 * 64 * 256 * 4);
  float*  kv_f   = (float*) alloc((size_t)16 * 256 * 4);
  float*  ret_f  = (float*) alloc((size_t)NP * 256 * 4);
  bf16_t* ret_b  = (bf16_t*)alloc((size_t)NP * 256 * 2);
  bf16_t* mid_b  = (bf16_t*)alloc((size_t)NP * 256 * 2);
  bf16_t* Bw_qkv[2], *Bw_o1[2], *Bw_o2[2];
  for (int a = 0; a < 2; ++a) {
    Bw_qkv[a] = (bf16_t*)alloc((size_t)256 * 768 * 2);
    Bw_o1[a]  = (bf16_t*)alloc((size_t)256 * 256 * 2);
    Bw_o2[a]  = (bf16_t*)alloc((size_t)256 * 256 * 2);
  }
  bf16_t* Bw_fc1 = (bf16_t*)alloc((size_t)256 * 256 * 2);

  dim3 blk(256);
  // ---- weight prep (f32 -> bf16 W^T, K padded) ----
  prep_dense<<<DIVUP(320 * 256, 256), blk, 0, stream>>>(c00_w, Bw_c00, 266, 320, 256);
  prep_dense<<<DIVUP(256 * 256, 256), blk, 0, stream>>>(fc1_w, Bw_fc1, 256, 256, 256);
  for (int a = 0; a < 2; ++a) {
    prep_dense<<<DIVUP(256 * 768, 256), blk, 0, stream>>>(attn[a][0], Bw_qkv[a], 256, 256, 768);
    prep_dense<<<DIVUP(256 * 256, 256), blk, 0, stream>>>(attn[a][6], Bw_o1[a], 256, 256, 256);
    prep_dense<<<DIVUP(256 * 256, 256), blk, 0, stream>>>(attn[a][8], Bw_o2[a], 256, 256, 256);
  }
  for (int i = 0; i < 16; ++i) {
    prep_conv<<<DIVUP(576 * 64, 256), blk, 0, stream>>>(res_w1 + (size_t)i * 576 * 64, Bw_r1 + (size_t)i * 576 * 64);
    prep_conv<<<DIVUP(576 * 64, 256), blk, 0, stream>>>(res_w2 + (size_t)i * 576 * 64, Bw_r2 + (size_t)i * 576 * 64);
  }
  prep_conv<<<DIVUP(576 * 64, 256), blk, 0, stream>>>(tail_w, Bw_tl);

  // ---- EDSR backbone ----
  head_conv<<<DIVUP(FP * 64, 256), blk, 0, stream>>>(inp, head_w, head_b, h_f, y_f, y_b);
  for (int i = 0; i < 16; ++i) {
    // z = relu(conv1(y) + b1)   (bf16 only)
    conv3_wmma<<<dim3(FP / 128, 1), blk, 0, stream>>>(
        y_b, Bw_r1 + (size_t)i * 576 * 64, res_b1 + i * 64,
        nullptr, nullptr, z_b, 2);
    // y = y + conv2(z) + b2     (f32 master + bf16 copy)
    conv3_wmma<<<dim3(FP / 128, 1), blk, 0, stream>>>(
        z_b, Bw_r2 + (size_t)i * 576 * 64, res_b2 + i * 64,
        y_f, y_f, y_b, 0);
  }
  // feat = conv_tail(y) + b + h
  conv3_wmma<<<dim3(FP / 128, 1), blk, 0, stream>>>(
      y_b, Bw_tl, tail_b, h_f, feat_f, nullptr, 0);

  // ---- grid + conv00 ----
  grid_assemble<<<NP / 8, blk, 0, stream>>>(feat_f, grid_b);
  gemm_dense<<<dim3(NP / 128, 256 / 128), blk, 0, stream>>>(
      grid_b, Bw_c00, c00_b, nullptr, x0_f, x0_b, NP, 256, 320, 0);

  // ---- two galerkin attention blocks ----
  float*  xf = x0_f; bf16_t* xb = x0_b;
  float*  of = x1_f; bf16_t* ob = x1_b;
  for (int a = 0; a < 2; ++a) {
    gemm_dense<<<dim3(NP / 128, 768 / 128), blk, 0, stream>>>(
        xb, Bw_qkv[a], attn[a][1], nullptr, nullptr, qkv_b, NP, 768, 256, 0);
    ln_kv<<<DIVUP(NP * 16, 256), blk, 0, stream>>>(
        qkv_b, attn[a][2], attn[a][3], attn[a][4], attn[a][5], kln_b, vln_b);
    kv_partial<<<16 * 64, blk, 0, stream>>>(kln_b, vln_b, kv_p);
    kv_finalize<<<16, blk, 0, stream>>>(kv_p, kv_f);
    attn_apply<<<NP * 16 / 256, blk, 0, stream>>>(qkv_b, kv_f, xf, ret_f, ret_b);
    // mid = gelu(o1(ret)+b)
    gemm_dense<<<dim3(NP / 128, 256 / 128), blk, 0, stream>>>(
        ret_b, Bw_o1[a], attn[a][7], nullptr, nullptr, mid_b, NP, 256, 256, 1);
    // out = o2(mid)+b + x
    gemm_dense<<<dim3(NP / 128, 256 / 128), blk, 0, stream>>>(
        mid_b, Bw_o2[a], attn[a][9], xf, (a == 0) ? of : nullptr, ob, NP, 256, 256, 0);
    float* tf = xf; xf = of; of = tf;
    bf16_t* tb = xb; xb = ob; ob = tb;
  }

  // ---- decoder: fc1 + gelu, then fc2 + bilinear skip ----
  gemm_dense<<<dim3(NP / 128, 256 / 128), blk, 0, stream>>>(
      xb, Bw_fc1, fc1_b, nullptr, nullptr, mid_b, NP, 256, 256, 1);
  final_fc2<<<NP / 8, blk, 0, stream>>>(mid_b, fc2_w, fc2_b, inp, out);
}
